// BiLSTM_SOFT_Encoder_42846593745063
// MI455X (gfx1250) — compile-verified, matrix operand order checked by reading, not run on hardware
//
#include <hip/hip_runtime.h>
#include <cstdint>
#include <cstddef>

// ---------------------------------------------------------------------------
// Types for CDNA5 WMMA (wave32): D(16x16 f32) = A(16x32 bf16) x B(32x16 bf16) + C
// ---------------------------------------------------------------------------
typedef __attribute__((ext_vector_type(16))) __bf16 bf16x16;
typedef __attribute__((ext_vector_type(8)))  float  f32x8;

#define WMMA_BF16(a, b, c) \
  __builtin_amdgcn_wmma_f32_16x16x32_bf16(false, (a), false, (b), (short)0, (c), false, false)

// CDNA5 async global->LDS path (ASYNCcnt). Builtin signature (from hipcc
// diagnostics): (v4i32 addrspace(1)*, v4i32 addrspace(3)*, imm offset, imm cpol)
#if __has_builtin(__builtin_amdgcn_global_load_async_to_lds_b128) && \
    __has_builtin(__builtin_amdgcn_s_wait_asynccnt)
#define ASYNC_LDS 1
typedef __attribute__((ext_vector_type(4))) int i32x4_t;
typedef __attribute__((address_space(1))) i32x4_t g_i32x4;
typedef __attribute__((address_space(3))) i32x4_t l_i32x4;
#else
#define ASYNC_LDS 0
#endif

static __device__ __forceinline__ uint16_t f2bf(float f) {
  uint32_t x = __builtin_bit_cast(uint32_t, f);
  x += 0x7FFFu + ((x >> 16) & 1u);   // round-to-nearest-even
  return (uint16_t)(x >> 16);
}
static __device__ __forceinline__ float sigmf(float x) { return 1.0f / (1.0f + __expf(-x)); }
static __device__ __forceinline__ f32x8 zero8() {
  f32x8 z = {0.f, 0.f, 0.f, 0.f, 0.f, 0.f, 0.f, 0.f};
  return z;
}

union FragU { bf16x16 v; uint4 u[2]; };

// Load a 16x32 bf16 fragment (A operand; also B operand of X @ W^T since W is
// row-major [N,K]).  ISA layout: lane&15 = matrix row; lanes 16-31 hold the
// K+8 halves; vgprs 0-3 = K[koff..koff+7], vgprs 4-7 = K[16+koff..16+koff+7].
static __device__ __forceinline__ bf16x16 load_frag(const uint16_t* base, int row, int ld,
                                                    int kbase, int lane) {
  int r    = row + (lane & 15);
  int koff = (lane >> 4) << 3;                // 0 or 8
  const uint16_t* p = base + (size_t)r * ld + kbase + koff;
  FragU f;
  f.u[0] = *(const uint4*)(p);                // 8 bf16, K = kbase+koff ..
  f.u[1] = *(const uint4*)(p + 16);           // 8 bf16, K = kbase+16+koff ..
  return f.v;
}

// ---------------------------------------------------------------------------
// Elementwise converters
// ---------------------------------------------------------------------------
__global__ void k_f32_to_bf16(const float* __restrict__ s, uint16_t* __restrict__ d, int n) {
  int i = blockIdx.x * 256 + threadIdx.x;
  if (i < n) d[i] = f2bf(s[i]);
}

// src: [mats, rows, cols] f32 (output-dim-last) -> dst: [mats, cols, rows] bf16 ("W-form")
__global__ void k_f32_to_bf16_T(const float* __restrict__ s, uint16_t* __restrict__ d,
                                int mats, int rows, int cols) {
  int i = blockIdx.x * 256 + threadIdx.x;
  int tot = mats * rows * cols;
  if (i >= tot) return;
  int m   = i / (rows * cols);
  int rem = i - m * rows * cols;
  int r   = rem / cols;
  int c   = rem - r * cols;
  d[(size_t)m * rows * cols + (size_t)c * rows + r] = f2bf(s[i]);
}

// s_term = sigmoid(xp[2] + hs0 + b[3]) * hs1   (fully precomputable)
__global__ void k_st(const float* __restrict__ xp5, const float* __restrict__ hs,
                     const float* __restrict__ b, float* __restrict__ st, int nrows) {
  int i = blockIdx.x * 256 + threadIdx.x;
  if (i >= nrows * 512) return;
  int row = i >> 9;
  int h   = i & 511;
  float g = sigmf(xp5[(size_t)row * 2560 + 1024 + h] + hs[(size_t)row * 1024 + h] + b[1536 + h]);
  st[i] = g * hs[(size_t)row * 1024 + 512 + h];
}

// ---------------------------------------------------------------------------
// bf16 WMMA GEMM:  C[M,N] = A[M,K] @ W[N,K]^T (+ bias0[n] + bias1[n])
// block = 256 threads = 8 waves, block tile 32M x 256N, wave tile 16M x 64N.
// Ping-pong K pipelining (2x unroll, disjoint fragment sets -> no reg copies).
// K must be a multiple of 64 (all K here are 512 or 1024).
// ---------------------------------------------------------------------------
__global__ __launch_bounds__(256) void gemm_bf16_wT(
    const uint16_t* __restrict__ A, const uint16_t* __restrict__ W,
    float* __restrict__ C, int M, int N, int K,
    const float* __restrict__ bias0, const float* __restrict__ bias1) {
  int lane = threadIdx.x & 31;
  int wave = threadIdx.x >> 5;                           // 0..7
  int mrow = blockIdx.y * 32 + (wave >> 2) * 16;
  int ncol = blockIdx.x * 256 + (wave & 3) * 64;
  (void)M;

  f32x8 acc[4];
#pragma unroll
  for (int j = 0; j < 4; ++j) acc[j] = zero8();

  bf16x16 aA = load_frag(A, mrow, K, 0, lane);
  bf16x16 bA[4];
#pragma unroll
  for (int j = 0; j < 4; ++j) bA[j] = load_frag(W, ncol + j * 16, K, 0, lane);

  for (int k = 0; k + 64 < K; k += 64) {
    __builtin_prefetch(W + (size_t)(ncol + (lane & 15)) * K + k + 64, 0, 1);
    bf16x16 aB = load_frag(A, mrow, K, k + 32, lane);
    bf16x16 bB[4];
#pragma unroll
    for (int j = 0; j < 4; ++j) bB[j] = load_frag(W, ncol + j * 16, K, k + 32, lane);
#pragma unroll
    for (int j = 0; j < 4; ++j) acc[j] = WMMA_BF16(aA, bA[j], acc[j]);
    aA = load_frag(A, mrow, K, k + 64, lane);
#pragma unroll
    for (int j = 0; j < 4; ++j) bA[j] = load_frag(W, ncol + j * 16, K, k + 64, lane);
#pragma unroll
    for (int j = 0; j < 4; ++j) acc[j] = WMMA_BF16(aB, bB[j], acc[j]);
  }
  {                                                      // tail: last 64 of K
    bf16x16 aB = load_frag(A, mrow, K, K - 32, lane);
    bf16x16 bB[4];
#pragma unroll
    for (int j = 0; j < 4; ++j) bB[j] = load_frag(W, ncol + j * 16, K, K - 32, lane);
#pragma unroll
    for (int j = 0; j < 4; ++j) acc[j] = WMMA_BF16(aA, bA[j], acc[j]);
#pragma unroll
    for (int j = 0; j < 4; ++j) acc[j] = WMMA_BF16(aB, bB[j], acc[j]);
  }

  int rbase = mrow + ((lane >> 4) << 3);                 // M = vgpr + 8*(lane>=16)
  int c0    = ncol + (lane & 15);                        // N = lane&15
#pragma unroll
  for (int j = 0; j < 4; ++j) {
    int col  = c0 + j * 16;
    float bv = (bias0 ? bias0[col] : 0.f) + (bias1 ? bias1[col] : 0.f);
#pragma unroll
    for (int v = 0; v < 8; ++v)
      C[(size_t)(rbase + v) * N + col] = acc[j][v] + bv;
  }
}

// ---------------------------------------------------------------------------
// BiLSTM recurrence, one direction per block (gridDim.x = 2), 1024 threads.
// Per step: gates[32,2048] = h[32,512] @ whh[2048,512]^T + xp(t); wave w owns
// h-columns [16w,16w+16) and computes tiles for all 4 gates -> no cross-wave
// exchange for the nonlinearity. h kept as bf16 in LDS, c in registers.
// xp(t+1) is staged into LDS with GLOBAL_LOAD_ASYNC_TO_LDS_B128 (ASYNCcnt),
// overlapping the next step's WMMA k-loop; falls back to global_prefetch_b8.
// xp layout: [2(dir)][B*T rows (row = b*256+t)][2048],  out: [rows][1024] bf16
// ---------------------------------------------------------------------------
__global__ __launch_bounds__(1024) void bilstm_dir_kernel(
    const float* __restrict__ xp, const uint16_t* __restrict__ whh,
    const int* __restrict__ lengths, uint16_t* __restrict__ outbf) {
  const int  dir = blockIdx.x;
  const bool rev = (dir == 1);
  const float*    xpd = xp + (size_t)dir * 8192 * 2048;
  const uint16_t* wd  = whh + (size_t)dir * 2048 * 512;

  __shared__ uint16_t hbuf[32 * 512];                    // 32 KB
#if ASYNC_LDS
  __shared__ float xstage[32 * 2048];                    // 256 KB xp slab (one step)
#endif
  int tid = threadIdx.x, lane = tid & 31, wave = tid >> 5;
  for (int i = tid; i < 32 * 512; i += 1024) hbuf[i] = 0;

  int hcol = wave * 16 + (lane & 15);                    // this lane's h column
  int bhi  = (lane >> 4) << 3;                           // 0 or 8 (M offset)

  float cc[2][8], hh[2][8];
  int   blen[2][8];
#pragma unroll
  for (int mt = 0; mt < 2; ++mt)
#pragma unroll
    for (int v = 0; v < 8; ++v) {
      cc[mt][v] = 0.f; hh[mt][v] = 0.f;
      blen[mt][v] = lengths[mt * 16 + bhi + v];
    }

#if ASYNC_LDS
  {                                                      // stage xp for step s=0
    int t0 = rev ? 255 : 0;
#pragma unroll
    for (int i = 0; i < 16; ++i) {
      int c = tid + i * 1024;                            // 16B chunk id, 0..16383
      int b = c >> 9;                                    // 512 chunks per row
      const float* g = xpd + ((size_t)b * 256 + t0) * 2048 + (size_t)(c & 511) * 4;
      __builtin_amdgcn_global_load_async_to_lds_b128(
          (g_i32x4*)g, (l_i32x4*)(xstage + (size_t)c * 4), 0, 0);
    }
  }
#endif
  __syncthreads();

  for (int s = 0; s < 256; ++s) {
    int t = rev ? (255 - s) : s;

    f32x8 acc[2][4];
#pragma unroll
    for (int mt = 0; mt < 2; ++mt)
#pragma unroll
      for (int g = 0; g < 4; ++g) acc[mt][g] = zero8();

    // -------- h @ whh^T : ping-pong pipelined over K (512) --------
    bf16x16 a0A = load_frag(hbuf, 0, 512, 0, lane);      // b rows 0..15 (LDS)
    bf16x16 a1A = load_frag(hbuf, 16, 512, 0, lane);     // b rows 16..31
    bf16x16 bA[4];
#pragma unroll
    for (int g = 0; g < 4; ++g) bA[g] = load_frag(wd, g * 512 + wave * 16, 512, 0, lane);
    for (int k = 0; k + 64 < 512; k += 64) {
      bf16x16 a0B = load_frag(hbuf, 0, 512, k + 32, lane);
      bf16x16 a1B = load_frag(hbuf, 16, 512, k + 32, lane);
      bf16x16 bB[4];
#pragma unroll
      for (int g = 0; g < 4; ++g)
        bB[g] = load_frag(wd, g * 512 + wave * 16, 512, k + 32, lane);
#pragma unroll
      for (int g = 0; g < 4; ++g) {
        acc[0][g] = WMMA_BF16(a0A, bA[g], acc[0][g]);
        acc[1][g] = WMMA_BF16(a1A, bA[g], acc[1][g]);
      }
      a0A = load_frag(hbuf, 0, 512, k + 64, lane);
      a1A = load_frag(hbuf, 16, 512, k + 64, lane);
#pragma unroll
      for (int g = 0; g < 4; ++g)
        bA[g] = load_frag(wd, g * 512 + wave * 16, 512, k + 64, lane);
#pragma unroll
      for (int g = 0; g < 4; ++g) {
        acc[0][g] = WMMA_BF16(a0B, bB[g], acc[0][g]);
        acc[1][g] = WMMA_BF16(a1B, bB[g], acc[1][g]);
      }
    }
    {                                                    // tail: K = 448..511
      bf16x16 a0B = load_frag(hbuf, 0, 512, 480, lane);
      bf16x16 a1B = load_frag(hbuf, 16, 512, 480, lane);
      bf16x16 bB[4];
#pragma unroll
      for (int g = 0; g < 4; ++g)
        bB[g] = load_frag(wd, g * 512 + wave * 16, 512, 480, lane);
#pragma unroll
      for (int g = 0; g < 4; ++g) {
        acc[0][g] = WMMA_BF16(a0A, bA[g], acc[0][g]);
        acc[1][g] = WMMA_BF16(a1A, bA[g], acc[1][g]);
      }
#pragma unroll
      for (int g = 0; g < 4; ++g) {
        acc[0][g] = WMMA_BF16(a0B, bB[g], acc[0][g]);
        acc[1][g] = WMMA_BF16(a1B, bB[g], acc[1][g]);
      }
    }

#if ASYNC_LDS
    __builtin_amdgcn_s_wait_asynccnt(0);                 // own stage loads done
#endif
    __syncthreads();                                     // stage(t) visible to all

    uint16_t hstore[2][8];
#pragma unroll
    for (int mt = 0; mt < 2; ++mt)
#pragma unroll
      for (int v = 0; v < 8; ++v) {
        int b = mt * 16 + bhi + v;
        size_t row = (size_t)b * 256 + t;
#if ASYNC_LDS
        const float* xr = xstage + (size_t)b * 2048;
#else
        const float* xr = xpd + row * 2048;
#endif
        float gi = sigmf(xr[hcol]        + acc[mt][0][v]);
        float gf = sigmf(xr[512 + hcol]  + acc[mt][1][v]);
        float gg = tanhf(xr[1024 + hcol] + acc[mt][2][v]);
        float go = sigmf(xr[1536 + hcol] + acc[mt][3][v]);
        float cn = gf * cc[mt][v] + gi * gg;
        float hn = go * tanhf(cn);
        bool  m  = t < blen[mt][v];
        cc[mt][v] = m ? cn : cc[mt][v];
        float h2  = m ? hn : hh[mt][v];
        hh[mt][v] = h2;
        hstore[mt][v] = f2bf(h2);
        outbf[row * 1024 + dir * 512 + hcol] = f2bf(m ? hn : 0.f);
      }
    __syncthreads();                                     // done reading hbuf/stage
#pragma unroll
    for (int mt = 0; mt < 2; ++mt)
#pragma unroll
      for (int v = 0; v < 8; ++v)
        hbuf[(size_t)(mt * 16 + bhi + v) * 512 + hcol] = hstore[mt][v];

    if (s + 1 < 256) {                                   // stage / prefetch xp(t+1)
      int tn = rev ? (255 - (s + 1)) : (s + 1);
#if ASYNC_LDS
#pragma unroll
      for (int i = 0; i < 16; ++i) {
        int c = tid + i * 1024;
        int b = c >> 9;
        const float* g = xpd + ((size_t)b * 256 + tn) * 2048 + (size_t)(c & 511) * 4;
        __builtin_amdgcn_global_load_async_to_lds_b128(
            (g_i32x4*)g, (l_i32x4*)(xstage + (size_t)c * 4), 0, 0);
      }
#else
      int r = tid >> 5, sg = tid & 31;                   // 32 threads/row, 256B each
      __builtin_prefetch((const char*)(xpd + ((size_t)r * 256 + tn) * 2048) + sg * 256, 0, 1);
#endif
    }
    __syncthreads();                                     // h visible for next step
  }
}

// ---------------------------------------------------------------------------
// Gated LSTM recurrence, one head per block (gridDim.x = 2: meta, hyper).
// Per step: hp[5][32,512] = h @ WhT[2560,512]^T ; all 5 mats for a wave's
// h-slice live in that wave's accumulators. Writes fp32 [B,T,H] to d_out.
// Per-step xp/st slab (384 KB) exceeds LDS next to hbuf -> prefetch instead.
// ---------------------------------------------------------------------------
__global__ __launch_bounds__(1024) void gated_dir_kernel(
    const float* __restrict__ xp_m, const float* __restrict__ st_m,
    const uint16_t* __restrict__ wh_m, const float* __restrict__ b_m,
    const float* __restrict__ xp_n, const float* __restrict__ st_n,
    const uint16_t* __restrict__ wh_n, const float* __restrict__ b_n,
    float* __restrict__ out) {
  const int net = blockIdx.x;
  const float*    xp = net ? xp_n : xp_m;                // [8192][2560]
  const float*    st = net ? st_n : st_m;                // [8192][512]
  const uint16_t* wh = net ? wh_n : wh_m;                // [2560][512] bf16
  const float*    bb = net ? b_n : b_m;                  // [6][512]
  float* o = out + (size_t)net * 8192 * 512;

  __shared__ uint16_t hbuf[32 * 512];
  int tid = threadIdx.x, lane = tid & 31, wave = tid >> 5;
  for (int i = tid; i < 32 * 512; i += 1024) hbuf[i] = 0;

  int hcol = wave * 16 + (lane & 15);
  int bhi  = (lane >> 4) << 3;
  float bv0 = bb[hcol],            bv1 = bb[512 + hcol], bv2 = bb[1024 + hcol];
  float bv4 = bb[4 * 512 + hcol],  bv5 = bb[5 * 512 + hcol];

  float cc[2][8];
#pragma unroll
  for (int mt = 0; mt < 2; ++mt)
#pragma unroll
    for (int v = 0; v < 8; ++v) cc[mt][v] = 0.f;
  __syncthreads();

  for (int t = 0; t < 256; ++t) {
    f32x8 acc[2][5];
#pragma unroll
    for (int mt = 0; mt < 2; ++mt)
#pragma unroll
      for (int kk = 0; kk < 5; ++kk) acc[mt][kk] = zero8();

    // -------- h @ WhT : ping-pong pipelined over K (512) --------
    bf16x16 a0A = load_frag(hbuf, 0, 512, 0, lane);
    bf16x16 a1A = load_frag(hbuf, 16, 512, 0, lane);
    bf16x16 bA[5];
#pragma unroll
    for (int kk = 0; kk < 5; ++kk) bA[kk] = load_frag(wh, kk * 512 + wave * 16, 512, 0, lane);
    for (int k = 0; k + 64 < 512; k += 64) {
      bf16x16 a0B = load_frag(hbuf, 0, 512, k + 32, lane);
      bf16x16 a1B = load_frag(hbuf, 16, 512, k + 32, lane);
      bf16x16 bB[5];
#pragma unroll
      for (int kk = 0; kk < 5; ++kk)
        bB[kk] = load_frag(wh, kk * 512 + wave * 16, 512, k + 32, lane);
#pragma unroll
      for (int kk = 0; kk < 5; ++kk) {
        acc[0][kk] = WMMA_BF16(a0A, bA[kk], acc[0][kk]);
        acc[1][kk] = WMMA_BF16(a1A, bA[kk], acc[1][kk]);
      }
      a0A = load_frag(hbuf, 0, 512, k + 64, lane);
      a1A = load_frag(hbuf, 16, 512, k + 64, lane);
#pragma unroll
      for (int kk = 0; kk < 5; ++kk)
        bA[kk] = load_frag(wh, kk * 512 + wave * 16, 512, k + 64, lane);
#pragma unroll
      for (int kk = 0; kk < 5; ++kk) {
        acc[0][kk] = WMMA_BF16(a0B, bB[kk], acc[0][kk]);
        acc[1][kk] = WMMA_BF16(a1B, bB[kk], acc[1][kk]);
      }
    }
    {                                                    // tail: K = 448..511
      bf16x16 a0B = load_frag(hbuf, 0, 512, 480, lane);
      bf16x16 a1B = load_frag(hbuf, 16, 512, 480, lane);
      bf16x16 bB[5];
#pragma unroll
      for (int kk = 0; kk < 5; ++kk)
        bB[kk] = load_frag(wh, kk * 512 + wave * 16, 512, 480, lane);
#pragma unroll
      for (int kk = 0; kk < 5; ++kk) {
        acc[0][kk] = WMMA_BF16(a0A, bA[kk], acc[0][kk]);
        acc[1][kk] = WMMA_BF16(a1A, bA[kk], acc[1][kk]);
      }
#pragma unroll
      for (int kk = 0; kk < 5; ++kk) {
        acc[0][kk] = WMMA_BF16(a0B, bB[kk], acc[0][kk]);
        acc[1][kk] = WMMA_BF16(a1B, bB[kk], acc[1][kk]);
      }
    }

    uint16_t hstore[2][8];
#pragma unroll
    for (int mt = 0; mt < 2; ++mt)
#pragma unroll
      for (int v = 0; v < 8; ++v) {
        int b = mt * 16 + bhi + v;
        size_t row = (size_t)b * 256 + t;
        const float* xr = xp + row * 2560;
        float gi = sigmf(xr[hcol]             + acc[mt][0][v] + bv0);
        float gf = sigmf(xr[512 + hcol]       + acc[mt][1][v] + bv1);
        float gg = sigmf(xr[1024 + hcol]      + acc[mt][2][v] + bv2);
        float cs = tanhf(xr[1536 + hcol] + gg * acc[mt][3][v] + st[row * 512 + hcol] + bv4);
        float c  = gf * cc[mt][v] + gi * cs;
        cc[mt][v] = c;
        float go = sigmf(xr[2048 + hcol]      + acc[mt][4][v] + bv5);
        float h  = go * tanhf(c);
        o[row * 512 + hcol] = h;                          // [B,T,H] (row = b*T + t)
        hstore[mt][v] = f2bf(h);
      }
    __syncthreads();
#pragma unroll
    for (int mt = 0; mt < 2; ++mt)
#pragma unroll
      for (int v = 0; v < 8; ++v)
        hbuf[(size_t)(mt * 16 + bhi + v) * 512 + hcol] = hstore[mt][v];

    if (t + 1 < 256) {                                   // prefetch next-step slabs
      int r = tid >> 5, sg = tid & 31;                   // 32 threads per row
      const char* xb = (const char*)(xp + ((size_t)r * 256 + t + 1) * 2560) + sg * 320;
      __builtin_prefetch(xb, 0, 1);
      __builtin_prefetch(xb + 192, 0, 1);
      __builtin_prefetch((const char*)(st + ((size_t)r * 256 + t + 1) * 512) + sg * 64, 0, 1);
    }
    __syncthreads();
  }
}

// ---------------------------------------------------------------------------
// Host orchestration (graph-capture safe: no malloc/sync, everything on stream)
// ---------------------------------------------------------------------------
extern "C" void kernel_launch(void* const* d_in, const int* in_sizes, int n_in,
                              void* d_out, int out_size, void* d_ws, size_t ws_size,
                              hipStream_t stream) {
  (void)in_sizes; (void)n_in; (void)out_size; (void)ws_size;

  const float* inputs = (const float*)d_in[0];   // [32,256,512]
  const int*   lens   = (const int*)d_in[1];     // [32]
  const float* w_ih0  = (const float*)d_in[2];   // [2,2048,512]
  const float* w_hh0  = (const float*)d_in[3];   // [2,2048,512]
  const float* b_ih0  = (const float*)d_in[4];   // [2,2048]
  const float* b_hh0  = (const float*)d_in[5];
  const float* w_ih1  = (const float*)d_in[6];   // [2,2048,1024]
  const float* w_hh1  = (const float*)d_in[7];   // [2,2048,512]
  const float* b_ih1  = (const float*)d_in[8];
  const float* b_hh1  = (const float*)d_in[9];
  const float* m_Wi   = (const float*)d_in[10];  // [5,512,512]
  const float* m_Wh   = (const float*)d_in[11];  // [5,512,512]
  const float* m_Whs  = (const float*)d_in[12];  // [2,1024,512]
  const float* m_b    = (const float*)d_in[13];  // [6,512]
  const float* n_Wi   = (const float*)d_in[14];
  const float* n_Wh   = (const float*)d_in[15];
  const float* n_Whs  = (const float*)d_in[16];
  const float* n_b    = (const float*)d_in[17];

  char* ws = (char*)d_ws;
  size_t off = 0;
  auto alloc = [&](size_t bytes) -> void* {
    void* p = ws + off;
    off = (off + bytes + 255) & ~(size_t)255;
    return p;
  };

  const int ROWS = 32 * 256;   // 8192, row = b*T + t everywhere

  uint16_t* wih0b  = (uint16_t*)alloc((size_t)2 * 2048 * 512 * 2);
  uint16_t* whh0b  = (uint16_t*)alloc((size_t)2 * 2048 * 512 * 2);
  uint16_t* wih1b  = (uint16_t*)alloc((size_t)2 * 2048 * 1024 * 2);
  uint16_t* whh1b  = (uint16_t*)alloc((size_t)2 * 2048 * 512 * 2);
  uint16_t* mWiT   = (uint16_t*)alloc((size_t)5 * 512 * 512 * 2);
  uint16_t* mWhT   = (uint16_t*)alloc((size_t)5 * 512 * 512 * 2);
  uint16_t* mWhsT  = (uint16_t*)alloc((size_t)2 * 1024 * 512 * 2);
  uint16_t* nWiT   = (uint16_t*)alloc((size_t)5 * 512 * 512 * 2);
  uint16_t* nWhT   = (uint16_t*)alloc((size_t)5 * 512 * 512 * 2);
  uint16_t* nWhsT  = (uint16_t*)alloc((size_t)2 * 1024 * 512 * 2);
  uint16_t* xbf    = (uint16_t*)alloc((size_t)ROWS * 512 * 2);
  uint16_t* out0b  = (uint16_t*)alloc((size_t)ROWS * 1024 * 2);
  uint16_t* outbib = (uint16_t*)alloc((size_t)ROWS * 1024 * 2);
  float*    xpL    = (float*)alloc((size_t)2 * ROWS * 2048 * 4);  // reused as xp5_m
  float*    xp5m   = xpL;                                         // [ROWS][2560]
  float*    xp5n   = (float*)alloc((size_t)ROWS * 2560 * 4);
  float*    hsbuf  = (float*)alloc((size_t)ROWS * 1024 * 4);
  float*    stm    = (float*)alloc((size_t)ROWS * 512 * 4);
  float*    stn    = (float*)alloc((size_t)ROWS * 512 * 4);

  auto cvt = [&](const float* s, uint16_t* d, int n) {
    k_f32_to_bf16<<<(n + 255) / 256, 256, 0, stream>>>(s, d, n);
  };
  auto cvtT = [&](const float* s, uint16_t* d, int mats, int rows, int cols) {
    int n = mats * rows * cols;
    k_f32_to_bf16_T<<<(n + 255) / 256, 256, 0, stream>>>(s, d, mats, rows, cols);
  };
  auto gemm = [&](const uint16_t* A, const uint16_t* W, float* C, int M, int N, int K,
                  const float* bias0, const float* bias1) {
    gemm_bf16_wT<<<dim3(N / 256, M / 32), 256, 0, stream>>>(A, W, C, M, N, K, bias0, bias1);
  };

  // Phase 0: weight/input conversion (and W-form transposes for the gated heads)
  cvt(w_ih0, wih0b, 2 * 2048 * 512);
  cvt(w_hh0, whh0b, 2 * 2048 * 512);
  cvt(w_ih1, wih1b, 2 * 2048 * 1024);
  cvt(w_hh1, whh1b, 2 * 2048 * 512);
  cvt(inputs, xbf, ROWS * 512);
  cvtT(m_Wi, mWiT, 5, 512, 512);
  cvtT(m_Wh, mWhT, 5, 512, 512);
  cvtT(m_Whs, mWhsT, 2, 1024, 512);
  cvtT(n_Wi, nWiT, 5, 512, 512);
  cvtT(n_Wh, nWhT, 5, 512, 512);
  cvtT(n_Whs, nWhsT, 2, 1024, 512);

  // Phase 1: BiLSTM layer 0
  for (int d = 0; d < 2; ++d)
    gemm(xbf, wih0b + (size_t)d * 2048 * 512, xpL + (size_t)d * ROWS * 2048,
         ROWS, 2048, 512, b_ih0 + d * 2048, b_hh0 + d * 2048);
  bilstm_dir_kernel<<<2, 1024, 0, stream>>>(xpL, whh0b, lens, out0b);

  // Phase 2: BiLSTM layer 1 (input = layer0 output, 1024 wide)
  for (int d = 0; d < 2; ++d)
    gemm(out0b, wih1b + (size_t)d * 2048 * 1024, xpL + (size_t)d * ROWS * 2048,
         ROWS, 2048, 1024, b_ih1 + d * 2048, b_hh1 + d * 2048);
  bilstm_dir_kernel<<<2, 1024, 0, stream>>>(xpL, whh1b, lens, outbib);

  // Phase 3: gated-head precomputation (xp5, hs, s_term) — m then n (hs reused)
  gemm(xbf, mWiT, xp5m, ROWS, 2560, 512, nullptr, nullptr);
  gemm(outbib, mWhsT, hsbuf, ROWS, 1024, 1024, nullptr, nullptr);
  k_st<<<(ROWS * 512 + 255) / 256, 256, 0, stream>>>(xp5m, hsbuf, m_b, stm, ROWS);
  gemm(xbf, nWiT, xp5n, ROWS, 2560, 512, nullptr, nullptr);
  gemm(outbib, nWhsT, hsbuf, ROWS, 1024, 1024, nullptr, nullptr);
  k_st<<<(ROWS * 512 + 255) / 256, 256, 0, stream>>>(xp5n, hsbuf, n_b, stn, ROWS);

  // Phase 4: both gated recurrences concurrently; writes meta|hyper to d_out
  gated_dir_kernel<<<2, 1024, 0, stream>>>(xp5m, stm, mWhT, m_b,
                                           xp5n, stn, nWhT, n_b, (float*)d_out);
}
